// DeepGeneNet_16587163697946
// MI455X (gfx1250) — compile-verified, matrix operand order checked by reading, not run on hardware
//
#include <hip/hip_runtime.h>
#include <hip/hip_bf16.h>

typedef __attribute__((ext_vector_type(16))) __bf16   bf16x16;
typedef __attribute__((ext_vector_type(8)))  __bf16   bf16x8;
typedef __attribute__((ext_vector_type(8)))  float    f32x8;
typedef __attribute__((ext_vector_type(4)))  unsigned u32x4;
typedef __attribute__((ext_vector_type(8)))  int      i32x8;
typedef __attribute__((ext_vector_type(4)))  int      i32x4;

#define NN   4096
#define DIN  768
#define H1   384
#define H2   256
#define H3   512
#define DOUT 128
#define EDGES 8192
#define KMID 129
#define ALPHA 0.2f
#define NEGV -9000000000000000.0f

#define TILE_ROWS   128           // B^T rows (output columns n)
#define TILE_K      64            // contiguous K per row
#define TILE_STRIDE 72            // LDS row stride in elements (64 + 8 pad = 144B = 9*16B)
#define TILE_BYTES  (TILE_ROWS * TILE_STRIDE * 2)

__device__ __forceinline__ float eluf(float x) { return x > 0.f ? x : (__expf(x) - 1.f); }
__device__ __forceinline__ float act_eval(float x, int kind) {
    if (kind == 0) return x > 0.f ? x : ALPHA * x;   // leaky_relu(0.2)
    return x > 0.f ? x : (__expf(x) - 1.f);          // elu
}

// ---------------------------------------------------------------- TDM: 2D tile (64 K x 128 rows, bf16) -> LDS
// D# padding: +16B every 128B  => LDS row stride 72 elements (16B-aligned rows, conflict-spread banks)
#if defined(__clang_major__) && (__clang_major__ >= 23)
#define TDM_CALL(g0, g1, z4a, z4b) \
    __builtin_amdgcn_tensor_load_to_lds(g0, g1, z4a, z4b, (i32x8){0,0,0,0,0,0,0,0}, 0)
#else
#define TDM_CALL(g0, g1, z4a, z4b) \
    __builtin_amdgcn_tensor_load_to_lds(g0, g1, z4a, z4b, 0)
#endif

__device__ __forceinline__ void tdm_issue_tile_64x128(unsigned lds_byte, const void* gsrc,
                                                      unsigned stride_elems) {
    unsigned long long ga = (unsigned long long)(size_t)gsrc;
    u32x4 g0;
    g0[0] = 1u;                                                    // count=1, user descriptor
    g0[1] = lds_byte;                                              // lds_addr (bytes)
    g0[2] = (unsigned)ga;                                          // global_addr[31:0]
    g0[3] = ((unsigned)(ga >> 32) & 0x01FFFFFFu) | (2u << 30);     // global_addr[56:32] | type=2
    i32x8 g1;
    g1[0] = (int)((1u << 16)        // data_size = 2 bytes
                | (1u << 20)        // pad_enable
                | (4u << 22)        // pad_interval: every 32 dwords (128B)
                | (3u << 25));      // pad_amount: 4 dwords (16B)
    g1[1] = (int)0xFFFF0000u;                     // tensor_dim0 = 0x7FFFFFFF (lo16 in [31:16])
    g1[2] = (int)0xFFFF7FFFu;                     // tensor_dim0 hi16 | tensor_dim1 lo16
    g1[3] = (int)(((unsigned)TILE_K << 16) | 0x7FFFu); // tensor_dim1 hi16 | tile_dim0 = 64
    g1[4] = TILE_ROWS;                            // tile_dim1 = 128 rows (tile_dim2 = 0)
    g1[5] = (int)stride_elems;                    // tensor_dim0_stride[31:0]
    g1[6] = 0;
    g1[7] = 0;
    i32x4 z4 = {0, 0, 0, 0};
    TDM_CALL(g0, g1, z4, z4);
}

// ---------------------------------------------------------------- cvt f32->bf16 (row-major keep)
__global__ void k_cvt4(const float* __restrict__ s, __bf16* __restrict__ d, int n) {
    int i = (blockIdx.x * 256 + threadIdx.x) * 4;
    if (i + 3 < n) {
        float4 v = *(const float4*)(s + i);
        d[i + 0] = (__bf16)v.x; d[i + 1] = (__bf16)v.y;
        d[i + 2] = (__bf16)v.z; d[i + 3] = (__bf16)v.w;
    }
}

// ---------------------------------------------------------------- cvt f32 [K][N] -> bf16 transposed [N][K]
__global__ void k_cvt_tr(const float* __restrict__ s, __bf16* __restrict__ d, int K, int N) {
    __shared__ float tile[32][33];
    int t = threadIdx.x;
    int tx = t & 31, ty = t >> 5;           // 32 x 8
    int n0 = blockIdx.x * 32, k0 = blockIdx.y * 32;
    #pragma unroll
    for (int i = 0; i < 4; ++i) {
        int k = ty * 4 + i;
        tile[k][tx] = s[(long long)(k0 + k) * N + n0 + tx];
    }
    __syncthreads();
    #pragma unroll
    for (int i = 0; i < 4; ++i) {
        int n = ty * 4 + i;
        d[(long long)(n0 + n) * K + k0 + tx] = (__bf16)tile[tx][n];
    }
}

// ---------------------------------------------------------------- adj -> bitmask
__global__ void k_pack_adj(const int* __restrict__ adj, unsigned* __restrict__ bits) {
    __shared__ unsigned tmp[8];
    int t = threadIdx.x;
    if (t < 8) tmp[t] = 0u;
    __syncthreads();
    int idx = blockIdx.x * 256 + t;
    unsigned bit = adj[idx] > 0 ? 1u : 0u;
    atomicOr(&tmp[t >> 5], bit << (t & 31));
    __syncthreads();
    if ((t & 31) == 0) bits[blockIdx.x * 8 + (t >> 5)] = tmp[t >> 5];
}

// ---------------------------------------------------------------- s1/s2 = h @ a[:H], h @ a[H:]
__global__ void k_rowdot2(const float* __restrict__ h, const float* __restrict__ a, int H,
                          float* __restrict__ s1, float* __restrict__ s2) {
    __shared__ float r1[256], r2[256];
    int i = blockIdx.x, t = threadIdx.x;
    float p1 = 0.f, p2 = 0.f;
    for (int c = t; c < H; c += 256) {
        float hv = h[(long long)i * H + c];
        p1 += hv * a[c];
        p2 += hv * a[H + c];
    }
    r1[t] = p1; r2[t] = p2; __syncthreads();
    for (int s = 128; s > 0; s >>= 1) {
        if (t < s) { r1[t] += r1[t + s]; r2[t] += r2[t + s]; }
        __syncthreads();
    }
    if (t == 0) { s1[i] = r1[0]; s2[i] = r2[0]; }
}

// ---------------------------------------------------------------- per-row softmax stats (max, sum)
__global__ void k_att_stats(const unsigned* __restrict__ bits, const float* __restrict__ s1,
                            const float* __restrict__ s2, float* __restrict__ mrow,
                            float* __restrict__ zrow, int kind) {
    __shared__ float s2s[NN];
    __shared__ unsigned mw[NN / 32];
    __shared__ float red[256];
    int i = blockIdx.x, t = threadIdx.x;
    for (int j = t; j < NN; j += 256) s2s[j] = s2[j];
    for (int w = t; w < NN / 32; w += 256) mw[w] = bits[i * (NN / 32) + w];
    __syncthreads();
    float s1i = s1[i];
    float mx = -3.4e38f;
    for (int j = t; j < NN; j += 256) {
        float e = ((mw[j >> 5] >> (j & 31)) & 1u) ? act_eval(s1i + s2s[j], kind) : NEGV;
        mx = fmaxf(mx, e);
    }
    red[t] = mx; __syncthreads();
    for (int s = 128; s > 0; s >>= 1) { if (t < s) red[t] = fmaxf(red[t], red[t + s]); __syncthreads(); }
    float m = red[0];
    __syncthreads();
    float sum = 0.f;
    for (int j = t; j < NN; j += 256) {
        float e = ((mw[j >> 5] >> (j & 31)) & 1u) ? act_eval(s1i + s2s[j], kind) : NEGV;
        sum += __expf(e - m);
    }
    red[t] = sum; __syncthreads();
    for (int s = 128; s > 0; s >>= 1) { if (t < s) red[t] += red[t + s]; __syncthreads(); }
    if (t == 0) { mrow[i] = m; zrow[i] = red[0]; }
}

// ---------------------------------------------------------------- acc = att @ h  (A on the fly, double-buffered TDM B^T, WMMA)
// hBt is transposed: [H][NN]. grid: (H/128, NN/16); block 256 = 8 waves.
template <int H, int KIND, int ACC>
__global__ __launch_bounds__(256) void k_att_gemm(const unsigned* __restrict__ bits,
                                                  const float* __restrict__ s1v,
                                                  const float* __restrict__ s2v,
                                                  const float* __restrict__ mrow,
                                                  const float* __restrict__ zrow,
                                                  const __bf16* __restrict__ hBt,
                                                  float* __restrict__ acc) {
    __shared__ __bf16 btT[2][TILE_ROWS][TILE_STRIDE];
    __shared__ float s2s[2][64];
    int t = threadIdx.x;
    int wave = t >> 5, lane = t & 31;
    int li = lane & 15, g = lane >> 4;
    int n0 = blockIdx.x * 128;
    int m0 = blockIdx.y * 16;
    int row = m0 + li;
    if ((int)blockDim.x == 9999) { btT[0][0][0] = (__bf16)0.f; s2s[0][0] = 0.f; }
    float s1i = s1v[row], mi = mrow[row], rz = 1.0f / zrow[row];
    const unsigned* brow = bits + row * (NN / 32);
    int nn = wave * 16 + li;
    unsigned ldsb = (unsigned)(size_t)(void*)&btT[0][0][0];
    const __bf16* bbase = hBt + (size_t)n0 * NN;   // rows n0.., stride NN, cols j
    f32x8 c;
    #pragma unroll
    for (int r = 0; r < 8; ++r) c[r] = 0.f;

    // prologue: stage tile 0 / s2 block 0
    if (t < 32) tdm_issue_tile_64x128(ldsb, bbase, NN);
    else if (t >= 64 && t < 128) s2s[0][t - 64] = s2v[t - 64];

    for (int j0 = 0, cur = 0; j0 < NN; j0 += 64, cur ^= 1) {
        if (t < 32) __builtin_amdgcn_s_wait_tensorcnt(0);
        __syncthreads();                               // publish tile `cur`
        if (j0 + 64 < NN) {                            // prefetch next tile / s2 block
            if (t < 32) tdm_issue_tile_64x128(ldsb + (unsigned)((cur ^ 1) * TILE_BYTES),
                                              bbase + j0 + 64, NN);
            else if (t >= 64 && t < 128) s2s[cur ^ 1][t - 64] = s2v[j0 + 64 + (t - 64)];
        }
        unsigned w0 = brow[(j0 >> 5)];
        unsigned w1 = brow[(j0 >> 5) + 1];
        const __bf16* bb = &btT[cur][nn][0];
        const float* ss = &s2s[cur][0];
        #pragma unroll
        for (int half = 0; half < 2; ++half) {
            const int kb = half * 32;
            unsigned word = half ? w1 : w0;
            bf16x16 a;
            #pragma unroll
            for (int j = 0; j < 8; ++j) {
                int kl0 = g * 8 + j;
                int kl1 = 16 + g * 8 + j;
                float e0 = ((word >> kl0) & 1u) ? act_eval(s1i + ss[kb + kl0], KIND) : NEGV;
                float e1 = ((word >> kl1) & 1u) ? act_eval(s1i + ss[kb + kl1], KIND) : NEGV;
                a[j]     = (__bf16)(__expf(e0 - mi) * rz);
                a[8 + j] = (__bf16)(__expf(e1 - mi) * rz);
            }
            bf16x8 blo = *(const bf16x8*)(bb + kb + g * 8);
            bf16x8 bhi = *(const bf16x8*)(bb + kb + 16 + g * 8);
            bf16x16 b = __builtin_shufflevector(blo, bhi, 0, 1, 2, 3, 4, 5, 6, 7,
                                                8, 9, 10, 11, 12, 13, 14, 15);
            c = __builtin_amdgcn_wmma_f32_16x16x32_bf16(false, a, false, b, (short)0, c, false, false);
        }
    }
    #pragma unroll
    for (int r = 0; r < 8; ++r) {
        long long idx = (long long)(m0 + r + g * 8) * H + n0 + nn;
        if (ACC) acc[idx] += c[r]; else acc[idx] = c[r];
    }
}

// ---------------------------------------------------------------- generic bf16 GEMM, B^T layout, double-buffered TDM
// Bt is [Nd][K]. CBMODE: 0=no bf16 out, 1=row-major, 2=transposed [Nd][NN].
template <int K, int Nd, int BIAS, int ELU, int CBMODE>
__global__ __launch_bounds__(256) void k_gemm(const __bf16* __restrict__ A,
                                              const __bf16* __restrict__ Bt,
                                              const float* __restrict__ bias,
                                              float* __restrict__ Cf,
                                              __bf16* __restrict__ Cb) {
    __shared__ __bf16 btT[2][TILE_ROWS][TILE_STRIDE];
    int t = threadIdx.x;
    int wave = t >> 5, lane = t & 31;
    int li = lane & 15, g = lane >> 4;
    int n0 = blockIdx.x * 128;
    int m0 = blockIdx.y * 16;
    int row = m0 + li;
    if ((int)blockDim.x == 9999) btT[0][0][0] = (__bf16)0.f;
    int nn = wave * 16 + li;
    unsigned ldsb = (unsigned)(size_t)(void*)&btT[0][0][0];
    const __bf16* arow = A + (size_t)row * K;
    const __bf16* bbase = Bt + (size_t)n0 * K;     // rows n0.., stride K, cols k
    f32x8 c;
    #pragma unroll
    for (int r = 0; r < 8; ++r) c[r] = 0.f;

    if (t < 32) tdm_issue_tile_64x128(ldsb, bbase, K);

    for (int k0 = 0, cur = 0; k0 < K; k0 += 64, cur ^= 1) {
        if (t < 32) __builtin_amdgcn_s_wait_tensorcnt(0);
        __syncthreads();
        if ((k0 + 64 < K) && t < 32)
            tdm_issue_tile_64x128(ldsb + (unsigned)((cur ^ 1) * TILE_BYTES), bbase + k0 + 64, K);
        const __bf16* bb = &btT[cur][nn][0];
        #pragma unroll
        for (int half = 0; half < 2; ++half) {
            const int kb = half * 32;
            bf16x8 alo = *(const bf16x8*)(arow + k0 + kb + g * 8);
            bf16x8 ahi = *(const bf16x8*)(arow + k0 + kb + 16 + g * 8);
            bf16x16 a = __builtin_shufflevector(alo, ahi, 0, 1, 2, 3, 4, 5, 6, 7,
                                                8, 9, 10, 11, 12, 13, 14, 15);
            bf16x8 blo = *(const bf16x8*)(bb + kb + g * 8);
            bf16x8 bhi = *(const bf16x8*)(bb + kb + 16 + g * 8);
            bf16x16 b = __builtin_shufflevector(blo, bhi, 0, 1, 2, 3, 4, 5, 6, 7,
                                                8, 9, 10, 11, 12, 13, 14, 15);
            c = __builtin_amdgcn_wmma_f32_16x16x32_bf16(false, a, false, b, (short)0, c, false, false);
        }
    }
    #pragma unroll
    for (int r = 0; r < 8; ++r) {
        int mr = m0 + r + g * 8;
        int col = n0 + nn;
        float v = c[r];
        if (BIAS) v += bias[col];
        if (ELU) v = eluf(v);
        Cf[(long long)mr * Nd + col] = v;
        if (CBMODE == 1) Cb[(long long)mr * Nd + col] = (__bf16)v;
        if (CBMODE == 2) Cb[(long long)col * NN + mr] = (__bf16)v;   // transposed for B-role reuse
    }
}

// ---------------------------------------------------------------- GAT epilogue: blend, elu, l2norm, +b, accumulate
__global__ void k_gat_epi(const float* __restrict__ acc, const float* __restrict__ h,
                          const float* __restrict__ bvec, float* __restrict__ gsum, int first) {
    __shared__ float o[H1];
    __shared__ float red[256];
    int i = blockIdx.x, t = threadIdx.x;
    float ss = 0.f;
    for (int c = t; c < H1; c += 256) {
        long long idx = (long long)i * H1 + c;
        float v = eluf(0.99f * acc[idx] + 0.01f * h[idx]);
        o[c] = v; ss += v * v;
    }
    red[t] = ss; __syncthreads();
    for (int s = 128; s > 0; s >>= 1) { if (t < s) red[t] += red[t + s]; __syncthreads(); }
    float inv = 1.0f / fmaxf(sqrtf(red[0]), 1e-12f);
    for (int c = t; c < H1; c += 256) {
        long long idx = (long long)i * H1 + c;
        float v = o[c] * inv + bvec[c];
        gsum[idx] = first ? v : (gsum[idx] + v);
    }
}

__global__ void k_xfin(const float* __restrict__ gsum, float* __restrict__ x, int n) {
    int i = blockIdx.x * 256 + threadIdx.x;
    if (i < n) x[i] = eluf(gsum[i] * (1.0f / 3.0f));
}

// ---------------------------------------------------------------- conv(3x129) + elu + channel mix (hcb transposed)
__global__ void k_conv(const float* __restrict__ x, const float* __restrict__ wmid,
                       const float* __restrict__ bmid, const float* __restrict__ wcore,
                       const float* __restrict__ bcore, float* __restrict__ hc,
                       __bf16* __restrict__ hcbT) {
    __shared__ float xs[H1];
    __shared__ float wm[3 * KMID];
    int n = blockIdx.x, t = threadIdx.x;
    for (int c = t; c < H1; c += 256) xs[c] = x[(long long)n * H1 + c];
    for (int c = t; c < 3 * KMID; c += 256) wm[c] = wmid[c];
    __syncthreads();
    int w = t;  // 0..255 output positions
    float val = 0.f;
    for (int cc = 0; cc < 3; ++cc) {
        float mc = 0.f;
        const float* wp = &wm[cc * KMID];
        for (int k = 0; k < KMID; ++k) mc += xs[w + k] * wp[k];
        val += wcore[cc] * eluf(mc + bmid[cc]);
    }
    float hv = val + bcore[0];
    hc[(long long)n * H2 + w] = hv;
    hcbT[(long long)w * NN + n] = (__bf16)hv;   // transposed [H2][NN]
}

// ---------------------------------------------------------------- embed: blend, elu, l2norm, +core_bias
__global__ void k_embed(const float* __restrict__ hp, const float* __restrict__ hc,
                        const float* __restrict__ cb, float* __restrict__ emb,
                        __bf16* __restrict__ embB) {
    __shared__ float red[256];
    int i = blockIdx.x, t = threadIdx.x;
    long long idx = (long long)i * H2 + t;
    float v = eluf(0.5f * (hp[idx] * 0.25f) + 0.5f * hc[idx]);
    red[t] = v * v; __syncthreads();
    for (int s = 128; s > 0; s >>= 1) { if (t < s) red[t] += red[t + s]; __syncthreads(); }
    float inv = 1.0f / fmaxf(sqrtf(red[0]), 1e-12f);
    float o = v * inv + cb[t];
    emb[idx] = o;
    embB[idx] = (__bf16)o;     // row-major: A-role downstream
}

// ---------------------------------------------------------------- edge dot products
__global__ void k_pred(const float* __restrict__ tf, const float* __restrict__ tg,
                       const int* __restrict__ samp, float* __restrict__ out) {
    int e = blockIdx.x * 256 + threadIdx.x;
    if (e >= EDGES) return;
    int ia = samp[2 * e], ib = samp[2 * e + 1];
    const float4* pa = (const float4*)(tf + (long long)ia * DOUT);
    const float4* pb = (const float4*)(tg + (long long)ib * DOUT);
    float s = 0.f;
    for (int k = 0; k < DOUT / 4; ++k) {
        float4 a = pa[k], b = pb[k];
        s += a.x * b.x + a.y * b.y + a.z * b.z + a.w * b.w;
    }
    out[e] = s;
}

// ================================================================ launch
extern "C" void kernel_launch(void* const* d_in, const int* in_sizes, int n_in,
                              void* d_out, int out_size, void* d_ws, size_t ws_size,
                              hipStream_t stream) {
    const float* xs_in[3] = { (const float*)d_in[0], (const float*)d_in[1], (const float*)d_in[2] };
    const int*   adj      = (const int*)d_in[3];
    const int*   samp     = (const int*)d_in[4];
    const float* gat_W    = (const float*)d_in[5];
    const float* gat_a    = (const float*)d_in[6];
    const float* gat_b    = (const float*)d_in[7];
    const float* w_mid    = (const float*)d_in[8];
    const float* b_mid    = (const float*)d_in[9];
    const float* w_core   = (const float*)d_in[10];
    const float* b_core   = (const float*)d_in[11];
    const float* a_core   = (const float*)d_in[12];
    const float* core_bias= (const float*)d_in[13];
    const float* W_tf1    = (const float*)d_in[14];
    const float* b_tf1    = (const float*)d_in[15];
    const float* W_tf2    = (const float*)d_in[16];
    const float* b_tf2    = (const float*)d_in[17];
    const float* W_tg1    = (const float*)d_in[18];
    const float* b_tg1    = (const float*)d_in[19];
    const float* W_tg2    = (const float*)d_in[20];
    const float* b_tg2    = (const float*)d_in[21];
    float* out = (float*)d_out;

    char* wsb = (char*)d_ws;
    size_t off = 0;
    auto alloc = [&](size_t bytes) -> char* {
        char* p = wsb + off;
        off += (bytes + 255) & ~(size_t)255;
        return p;
    };
    __bf16* gatWbT = (__bf16*)alloc((size_t)3 * DIN * H1 * 2);   // per layer: [H1][DIN]
    __bf16* Wtf1bT = (__bf16*)alloc((size_t)H3 * H2 * 2);        // [H3][H2]
    __bf16* Wtf2bT = (__bf16*)alloc((size_t)DOUT * H3 * 2);      // [DOUT][H3]
    __bf16* Wtg1bT = (__bf16*)alloc((size_t)H3 * H2 * 2);
    __bf16* Wtg2bT = (__bf16*)alloc((size_t)DOUT * H3 * 2);
    unsigned* bits = (unsigned*)alloc((size_t)NN * (NN / 32) * 4);
    __bf16* xb     = (__bf16*)alloc((size_t)NN * DIN * 2);       // row-major (A-role)
    float* hf      = (float*)alloc((size_t)NN * H1 * 4);
    __bf16* hbT    = (__bf16*)alloc((size_t)H1 * NN * 2);        // transposed (B-role)
    float* s1b     = (float*)alloc((size_t)NN * 4);
    float* s2b     = (float*)alloc((size_t)NN * 4);
    float* mb_     = (float*)alloc((size_t)NN * 4);
    float* zb      = (float*)alloc((size_t)NN * 4);
    float* accb    = (float*)alloc((size_t)NN * H1 * 4);
    float* gsum    = (float*)alloc((size_t)NN * H1 * 4);
    float* xbuf    = (float*)alloc((size_t)NN * H1 * 4);
    float* hcore   = (float*)alloc((size_t)NN * H2 * 4);
    __bf16* hcorebT= (__bf16*)alloc((size_t)H2 * NN * 2);        // transposed (B-role)
    float* hpass   = (float*)alloc((size_t)NN * H2 * 4);
    float* embf    = (float*)alloc((size_t)NN * H2 * 4);
    __bf16* embb   = (__bf16*)alloc((size_t)NN * H2 * 2);        // row-major (A-role)
    float* t1f     = (float*)alloc((size_t)NN * H3 * 4);
    __bf16* t1b    = (__bf16*)alloc((size_t)NN * H3 * 2);        // row-major (A-role)
    float* tff     = (float*)alloc((size_t)NN * DOUT * 4);
    float* tgf     = (float*)alloc((size_t)NN * DOUT * 4);

    // weights -> bf16 transposed
    for (int l = 0; l < 3; ++l)
        k_cvt_tr<<<dim3(H1 / 32, DIN / 32), 256, 0, stream>>>(
            gat_W + (size_t)l * DIN * H1, gatWbT + (size_t)l * H1 * DIN, DIN, H1);
    k_cvt_tr<<<dim3(H3 / 32, H2 / 32), 256, 0, stream>>>(W_tf1, Wtf1bT, H2, H3);
    k_cvt_tr<<<dim3(DOUT / 32, H3 / 32), 256, 0, stream>>>(W_tf2, Wtf2bT, H3, DOUT);
    k_cvt_tr<<<dim3(H3 / 32, H2 / 32), 256, 0, stream>>>(W_tg1, Wtg1bT, H2, H3);
    k_cvt_tr<<<dim3(DOUT / 32, H3 / 32), 256, 0, stream>>>(W_tg2, Wtg2bT, H3, DOUT);

    // adjacency -> bitmask (2 MB, L2-resident)
    k_pack_adj<<<(NN * NN) / 256, 256, 0, stream>>>(adj, bits);

    // ---- 3 GAT layers ----
    for (int l = 0; l < 3; ++l) {
        k_cvt4<<<(NN * DIN / 4) / 256, 256, 0, stream>>>(xs_in[l], xb, NN * DIN);
        k_gemm<DIN, H1, 0, 0, 2><<<dim3(H1 / 128, NN / 16), 256, 0, stream>>>(
            xb, gatWbT + (size_t)l * H1 * DIN, nullptr, hf, hbT);
        k_rowdot2<<<NN, 256, 0, stream>>>(hf, gat_a + (size_t)l * 2 * H1, H1, s1b, s2b);
        k_att_stats<<<NN, 256, 0, stream>>>(bits, s1b, s2b, mb_, zb, 0);
        k_att_gemm<H1, 0, 0><<<dim3(H1 / 128, NN / 16), 256, 0, stream>>>(
            bits, s1b, s2b, mb_, zb, hbT, accb);
        k_gat_epi<<<NN, 256, 0, stream>>>(accb, hf, gat_b + (size_t)l * H1, gsum, l == 0);
    }
    k_xfin<<<(NN * H1) / 256, 256, 0, stream>>>(gsum, xbuf, NN * H1);

    // ---- conv mixer -> core features ----
    k_conv<<<NN, 256, 0, stream>>>(xbuf, w_mid, b_mid, w_core, b_core, hcore, hcorebT);

    // ---- 4 core attention heads ----
    for (int hd = 0; hd < 4; ++hd) {
        k_rowdot2<<<NN, 256, 0, stream>>>(hcore, a_core + (size_t)hd * 2 * H2, H2, s1b, s2b);
        k_att_stats<<<NN, 256, 0, stream>>>(bits, s1b, s2b, mb_, zb, 1);
        if (hd == 0)
            k_att_gemm<H2, 1, 0><<<dim3(H2 / 128, NN / 16), 256, 0, stream>>>(
                bits, s1b, s2b, mb_, zb, hcorebT, hpass);
        else
            k_att_gemm<H2, 1, 1><<<dim3(H2 / 128, NN / 16), 256, 0, stream>>>(
                bits, s1b, s2b, mb_, zb, hcorebT, hpass);
    }
    k_embed<<<NN, 256, 0, stream>>>(hpass, hcore, core_bias, embf, embb);

    // ---- tf / tg MLPs ----
    k_gemm<H2, H3, 1, 1, 1><<<dim3(H3 / 128, NN / 16), 256, 0, stream>>>(embb, Wtf1bT, b_tf1, t1f, t1b);
    k_gemm<H3, DOUT, 1, 1, 0><<<dim3(DOUT / 128, NN / 16), 256, 0, stream>>>(t1b, Wtf2bT, b_tf2, tff, (__bf16*)nullptr);
    k_gemm<H2, H3, 1, 1, 1><<<dim3(H3 / 128, NN / 16), 256, 0, stream>>>(embb, Wtg1bT, b_tg1, t1f, t1b);
    k_gemm<H3, DOUT, 1, 1, 0><<<dim3(DOUT / 128, NN / 16), 256, 0, stream>>>(t1b, Wtg2bT, b_tg2, tgf, (__bf16*)nullptr);

    // ---- edge predictions ----
    k_pred<<<EDGES / 256, 256, 0, stream>>>(tff, tgf, samp, out);
    (void)in_sizes; (void)n_in; (void)out_size; (void)ws_size;
}
